// BinLinear_14027363189197
// MI455X (gfx1250) — compile-verified, matrix operand order checked by reading, not run on hardware
//
#include <hip/hip_runtime.h>

// D = X @ sign(W):  X[8192,4096] f32, W[4096,4096] f32.
// sign(w) in {+1,-1} is exact in bf16; X is split hi+lo (both bf16) for ~fp32
// accuracy with f32 WMMA accumulation. Double-buffered LDS software pipeline.

typedef __attribute__((ext_vector_type(16))) __bf16 bf16x16;
typedef __attribute__((ext_vector_type(8)))  float  f32x8;

#define M_TOTAL 8192
#define N_TOTAL 4096
#define K_TOTAL 4096

#define BM 128                  // block tile M (4 M-waves * 32)
#define BN 128                  // block tile N (2 N-waves * 64)
#define BK 32                   // one bf16 WMMA K step
#define LDST 40                 // shorts/row: 32 data + 8 pad = 80B (16B-aligned, conflict-free reads)
#define KSTEPS (K_TOTAL / BK)   // 128

union Frag  { unsigned u[8]; bf16x16 v; };
union Pack2 { __bf16 h[2]; unsigned u; };

__device__ __forceinline__ unsigned pack_bf16x2(float a, float b) {
    Pack2 p; p.h[0] = (__bf16)a; p.h[1] = (__bf16)b;   // v_cvt_pk_bf16_f32 (RNE)
    return p.u;
}
__device__ __forceinline__ float bf16_round(float a) {
    return (float)((__bf16)a);
}
__device__ __forceinline__ unsigned bin_bf16(float w) {
    // reference: w >= 0 -> +1 (incl. -0.0), else -1
    return (w >= 0.f) ? 0x3F80u : 0xBF80u;
}

__global__ __launch_bounds__(256)
void binlinear_wmma_bf16split(const float* __restrict__ X,
                              const float* __restrict__ W,
                              float* __restrict__ Out) {
    // double-buffered LDS: A hi/lo planes row-major [M][K], binarized W transposed [N][K]
    __shared__ __align__(16) unsigned short Ahi[2][BM][LDST];
    __shared__ __align__(16) unsigned short Alo[2][BM][LDST];
    __shared__ __align__(16) unsigned short Bt [2][BN][LDST];   // 3*2*10240 = 60 KB

    const int tid  = threadIdx.x;
    const int lane = tid & 31;          // wave32
    const int wave = tid >> 5;          // 8 waves
    const int half = lane >> 4;         // K-half selector for 16-bit operand layout
    const int mr   = lane & 15;         // row (A) / col (B) within 16x16 tile
    const int mw   = wave >> 1;         // 0..3 : M wave slot (32 rows each)
    const int nw   = wave & 1;          // 0..1 : N wave slot (64 cols each)

    const int m_block = blockIdx.y * BM;
    const int n_block = blockIdx.x * BN;

    // X staging coords: 128x32 tile, float4 along K, 4 passes of 32 rows
    const int arow = tid >> 3;          // 0..31
    const int acol = (tid & 7) * 4;     // 0,4,..28
    // W staging coords: 32x128 tile, per-lane columns (coalesced scalar loads)
    const int wk    = (tid >> 5) * 4;   // 0,4,..28 : 4 consecutive K rows per wave
    const int wlane = tid & 31;         // n within 32-col group, 4 passes

    const f32x8 zero = {0.f,0.f,0.f,0.f,0.f,0.f,0.f,0.f};
    f32x8 acc[2][4];
    #pragma unroll
    for (int i = 0; i < 2; ++i)
        #pragma unroll
        for (int j = 0; j < 4; ++j) acc[i][j] = zero;

    float4 xr[4];
    float  wr[4][4];                    // [n-pass][k]

    auto load_tile = [&](int k0) {
        #pragma unroll
        for (int p = 0; p < 4; ++p)
            xr[p] = *(const float4*)(X + (size_t)(m_block + p * 32 + arow) * K_TOTAL + k0 + acol);
        #pragma unroll
        for (int np = 0; np < 4; ++np)
            #pragma unroll
            for (int q = 0; q < 4; ++q)
                wr[np][q] = W[(size_t)(k0 + wk + q) * N_TOTAL + n_block + np * 32 + wlane];
    };

    auto store_tile = [&](int buf) {
        #pragma unroll
        for (int p = 0; p < 4; ++p) {
            const int r = p * 32 + arow;
            const float a0 = xr[p].x, a1 = xr[p].y, a2 = xr[p].z, a3 = xr[p].w;
            uint2 hp, lp;
            hp.x = pack_bf16x2(a0, a1);
            hp.y = pack_bf16x2(a2, a3);
            lp.x = pack_bf16x2(a0 - bf16_round(a0), a1 - bf16_round(a1));
            lp.y = pack_bf16x2(a2 - bf16_round(a2), a3 - bf16_round(a3));
            *(uint2*)&Ahi[buf][r][acol] = hp;                   // ds_store_b64
            *(uint2*)&Alo[buf][r][acol] = lp;
        }
        #pragma unroll
        for (int np = 0; np < 4; ++np) {
            const unsigned s0 = bin_bf16(wr[np][0]);
            const unsigned s1 = bin_bf16(wr[np][1]);
            const unsigned s2 = bin_bf16(wr[np][2]);
            const unsigned s3 = bin_bf16(wr[np][3]);
            uint2 bp;
            bp.x = s0 | (s1 << 16);
            bp.y = s2 | (s3 << 16);
            *(uint2*)&Bt[buf][np * 32 + wlane][wk] = bp;        // transposed, conflict-free
        }
    };

    auto compute_tile = [&](int buf) {
        // fragment layout (ISA 16-bit operands): lane (half, mr);
        // VGPR0..3 <- K = half*8..+7 ; VGPR4..7 <- K = 16+half*8..+7
        Frag a_hi[2], a_lo[2], bfr[4];
        #pragma unroll
        for (int i = 0; i < 2; ++i) {
            const unsigned short* ah = &Ahi[buf][mw * 32 + i * 16 + mr][0];
            const unsigned short* al = &Alo[buf][mw * 32 + i * 16 + mr][0];
            *(uint4*)&a_hi[i].u[0] = *(const uint4*)(ah + half * 8);       // ds_load_b128
            *(uint4*)&a_hi[i].u[4] = *(const uint4*)(ah + 16 + half * 8);
            *(uint4*)&a_lo[i].u[0] = *(const uint4*)(al + half * 8);
            *(uint4*)&a_lo[i].u[4] = *(const uint4*)(al + 16 + half * 8);
        }
        #pragma unroll
        for (int j = 0; j < 4; ++j) {
            const unsigned short* bc = &Bt[buf][nw * 64 + j * 16 + mr][0];
            *(uint4*)&bfr[j].u[0] = *(const uint4*)(bc + half * 8);
            *(uint4*)&bfr[j].u[4] = *(const uint4*)(bc + 16 + half * 8);
        }
        #pragma unroll
        for (int i = 0; i < 2; ++i) {
            #pragma unroll
            for (int j = 0; j < 4; ++j) {
                acc[i][j] = __builtin_amdgcn_wmma_f32_16x16x32_bf16(
                    false, a_hi[i].v, false, bfr[j].v, (short)0, acc[i][j], false, false);
                acc[i][j] = __builtin_amdgcn_wmma_f32_16x16x32_bf16(
                    false, a_lo[i].v, false, bfr[j].v, (short)0, acc[i][j], false, false);
            }
        }
    };

    // ---- software pipeline: load(k+1) -> compute(k) -> convert/store(k+1) -> barrier ----
    load_tile(0);
    store_tile(0);
    __syncthreads();

    for (int kk = 0; kk < KSTEPS; ++kk) {
        const int cur = kk & 1;
        if (kk + 1 < KSTEPS) {
            load_tile((kk + 1) * BK);                        // global loads in flight
            if (kk + 2 < KSTEPS) {                           // L2 prefetch 2 tiles ahead
                __builtin_prefetch(X + (size_t)(m_block + arow) * K_TOTAL + (kk + 2) * BK + acol, 0, 1);
                __builtin_prefetch(W + (size_t)((kk + 2) * BK + wk) * N_TOTAL + n_block + wlane, 0, 1);
            }
        }
        compute_tile(cur);                                   // 16x v_wmma overlap the loads
        if (kk + 1 < KSTEPS) store_tile(cur ^ 1);
        __syncthreads();
    }

    // ---- epilogue: C VGPR r of lane (half, mr) = element (r + 8*half, mr) ----
    #pragma unroll
    for (int i = 0; i < 2; ++i) {
        #pragma unroll
        for (int j = 0; j < 4; ++j) {
            const int col = n_block + nw * 64 + j * 16 + mr;
            #pragma unroll
            for (int r = 0; r < 8; ++r) {
                const int row = m_block + mw * 32 + i * 16 + r + 8 * half;
                Out[(size_t)row * N_TOTAL + col] = acc[i][j][r];
            }
        }
    }
}

extern "C" void kernel_launch(void* const* d_in, const int* in_sizes, int n_in,
                              void* d_out, int out_size, void* d_ws, size_t ws_size,
                              hipStream_t stream) {
    const float* x = (const float*)d_in[0];   // [8192, 4096]
    const float* w = (const float*)d_in[1];   // [4096, 4096]
    float* out = (float*)d_out;               // [8192, 4096]
    (void)in_sizes; (void)n_in; (void)out_size; (void)d_ws; (void)ws_size;

    dim3 block(256, 1, 1);
    dim3 grid(N_TOTAL / BN, M_TOTAL / BM, 1); // 32 x 64 = 2048 workgroups
    binlinear_wmma_bf16split<<<grid, block, 0, stream>>>(x, w, out);
}
